// PointnetSAModule_5153960755817
// MI455X (gfx1250) — matrix operand order repaired and checked by reading, not compiled, not run on hardware
//
#include <hip/hip_runtime.h>

typedef _Float16 half_t;
typedef __attribute__((ext_vector_type(16))) _Float16 v16h;
typedef __attribute__((ext_vector_type(8)))  _Float16 v8h;
typedef __attribute__((ext_vector_type(8)))  float    v8f;

#define BATCH 8
#define NPTS  4096
#define CFEAT 128
#define SPTS  1024        // NPOINT
#define KS    64          // NSAMPLE
#define R2    0.04f       // RADIUS^2
#define BNEPS 1e-5f
#define NGROUP (BATCH*SPTS)            // 8192 (b,s) groups
#define MTOT  ((long long)NGROUP*KS)   // 524288 rows through the MLP

// ---- workspace layout (bytes) ----
static const size_t OFF_GI    = 0;                          // 8192*64 int = 2 MB
static const size_t OFF_STATS = (size_t)NGROUP*KS*4;        // 2048 floats = 8 KB
static const size_t OFF_WT1   = OFF_STATS + 8192;           // f16 W1^T [128][160] = 40960 B
static const size_t OFF_WT2   = OFF_WT1 + 128*160*2;        // f16 W2^T [128][128] = 32768 B
static const size_t OFF_WT3   = OFF_WT2 + 128*128*2;        // f16 W3^T [256][128] = 65536 B
static const size_t OFF_Y1    = OFF_WT3 + 256*128*2;        // f16 [524288 x 128]
static const size_t OFF_Y2    = OFF_Y1 + (size_t)NGROUP*KS*128*2;
static const size_t OFF_Y3    = OFF_Y2 + (size_t)NGROUP*KS*128*2;

// ---- CDNA5 async copy: global -> LDS, 16 bytes, tracked by ASYNCcnt ----
__device__ __forceinline__ void async_copy16(void* lds_dst, const void* gsrc) {
  asm volatile("global_load_async_to_lds_b128 %0, %1, off"
               :: "v"((unsigned)(size_t)lds_dst),
                  "v"((unsigned long long)(size_t)gsrc)
               : "memory");
}
__device__ __forceinline__ void wait_async0() {
  asm volatile("s_wait_asynccnt 0" ::: "memory");
}

// =====================================================================
// 1) Farthest-point sampling: one block per batch, sequential 1024 iters
// =====================================================================
__global__ __launch_bounds__(512) void fps_kernel(const float* __restrict__ xyz,
                                                  float* __restrict__ new_xyz) {
  __shared__ float sdist[NPTS];
  __shared__ float wval[16];
  __shared__ int   widx[16];
  __shared__ int   sfar;
  const int b = blockIdx.x, t = threadIdx.x;
  const float* base = xyz + (size_t)b * NPTS * 3;

  for (int i = t; i < NPTS; i += 512) sdist[i] = 1e10f;
  if (t == 0) sfar = 0;
  __syncthreads();

  int far = 0;
  for (int it = 0; it < SPTS; ++it) {
    const float px = base[3*far], py = base[3*far+1], pz = base[3*far+2];
    if (t == 0) {
      float* o = &new_xyz[((size_t)b * SPTS + it) * 3];
      o[0] = px; o[1] = py; o[2] = pz;
    }
    float best = -1.0f; int bidx = 0;
    for (int i = t; i < NPTS; i += 512) {
      const float dx = base[3*i] - px, dy = base[3*i+1] - py, dz = base[3*i+2] - pz;
      const float d = dx*dx + dy*dy + dz*dz;
      float m = sdist[i]; m = fminf(m, d); sdist[i] = m;
      if (m > best) { best = m; bidx = i; }
    }
    // wave32 reduction (argmax, prefer lower index on ties)
    for (int off = 16; off > 0; off >>= 1) {
      const float ov = __shfl_down(best, off, 32);
      const int   oi = __shfl_down(bidx, off, 32);
      if (ov > best || (ov == best && oi < bidx)) { best = ov; bidx = oi; }
    }
    const int lane = t & 31, wid = t >> 5;
    if (lane == 0) { wval[wid] = best; widx[wid] = bidx; }
    __syncthreads();
    if (t < 16) {
      best = wval[t]; bidx = widx[t];
      for (int off = 8; off > 0; off >>= 1) {
        const float ov = __shfl_down(best, off, 16);
        const int   oi = __shfl_down(bidx, off, 16);
        if (ov > best || (ov == best && oi < bidx)) { best = ov; bidx = oi; }
      }
      if (t == 0) sfar = bidx;
    }
    __syncthreads();
    far = sfar;
  }
}

// =====================================================================
// 2) Ball query: one wave per query; ordered compaction of first 64 hits
// =====================================================================
__global__ __launch_bounds__(256) void qbp_kernel(const float* __restrict__ xyz,
                                                  const float* __restrict__ new_xyz,
                                                  int* __restrict__ gi) {
  const int q = blockIdx.x * 8 + (threadIdx.x >> 5);
  const int lane = threadIdx.x & 31;
  if (q >= NGROUP) return;
  const int b = q >> 10;
  const float* c3 = new_xyz + (size_t)q * 3;
  const float qx = c3[0], qy = c3[1], qz = c3[2];
  const float* base = xyz + (size_t)b * NPTS * 3;
  int* out = gi + (size_t)q * KS;

  int cnt = 0;
  for (int i0 = 0; i0 < NPTS && cnt < KS; i0 += 32) {
    const int i = i0 + lane;
    const float dx = base[3*i] - qx, dy = base[3*i+1] - qy, dz = base[3*i+2] - qz;
    const float d = dx*dx + dy*dy + dz*dz;
    const bool pred = (d <= R2);
    const unsigned m = (unsigned)__ballot(pred);
    const int pre = __popc(m & ((1u << lane) - 1u));
    if (pred) { const int pos = cnt + pre; if (pos < KS) out[pos] = i; }
    cnt += __popc(m);
  }
  if (cnt < KS) {  // pad with first index (>=1 hit guaranteed: the center itself)
    const int first = out[0];
    for (int j = cnt + lane; j < KS; j += 32) out[j] = first;
  }
}

// =====================================================================
// 3) stats zero + W pre-transpose/convert + BN finalize
// =====================================================================
__global__ void zero_stats(float* p, int n) {
  const int i = blockIdx.x * blockDim.x + threadIdx.x;
  if (i < n) p[i] = 0.f;
}

// WT[c*KPAD + k] = (half)W[k*Ntot + c], zero-padded to KPAD
__global__ void prep_w(const float* __restrict__ W, half_t* __restrict__ WT,
                       int kreal, int kpad, int ntot) {
  const int idx = blockIdx.x * blockDim.x + threadIdx.x;
  if (idx >= ntot * kpad) return;
  const int c = idx / kpad, k = idx - c * kpad;
  WT[idx] = (k < kreal) ? (half_t)W[(size_t)k * ntot + c] : (half_t)0.f;
}

__global__ void bn_finalize(const float* __restrict__ sum, const float* __restrict__ sq,
                            const float* __restrict__ gg, const float* __restrict__ bb,
                            float* __restrict__ scale, float* __restrict__ shift, int nc) {
  const int c = blockIdx.x * blockDim.x + threadIdx.x;
  if (c >= nc) return;
  const float inv = 1.0f / (float)MTOT;
  const float m = sum[c] * inv;
  const float v = sq[c] * inv - m * m;
  const float sc = gg[c] * rsqrtf(v + BNEPS);
  scale[c] = sc;
  shift[c] = bb[c] - m * sc;
}

// =====================================================================
// 4) Fused (gather | BN+ReLU) -> f16 WMMA GEMM -> y(f16) + channel stats
//    block = one (b,s) group: M=64 rows; 8 waves, 4 WMMA tiles each.
//    W^T tile staged via async-to-LDS DMA overlapped with A staging.
// =====================================================================
template <int KPAD, int KREAL, bool GATHER>
__global__ __launch_bounds__(256) void gemm_bn_kernel(
    const float* __restrict__ xyz, const float* __restrict__ feat,
    const float* __restrict__ new_xyz, const int* __restrict__ gi,
    const half_t* __restrict__ yin, const float* __restrict__ scale,
    const float* __restrict__ shift, const half_t* __restrict__ WT, int NtotOut,
    half_t* __restrict__ yout, float* __restrict__ gsum, float* __restrict__ gsq) {
  __shared__ half_t lA[64 * KPAD];    // A tile, row-major [row][k]
  __shared__ half_t lW[128 * KPAD];   // W^T tile, [col][k]
  __shared__ float  lsum[128], lsq[128];
  __shared__ float  lscale[128], lshift[128];
  __shared__ int    pidx[64];
  __shared__ float  nx[3];

  const int g = blockIdx.x;
  const int n_off = blockIdx.y * 128;
  const int tid = threadIdx.x;
  const int b = g >> 10;

  // ---- kick off W^T tile DMA first: 16B chunks, ASYNCcnt-tracked ----
  {
    const half_t* wsrc = WT + (size_t)n_off * KPAD;
    constexpr int NCH = 128 * KPAD / 8;   // # of 16-byte chunks
    #pragma unroll
    for (int idx = tid; idx < NCH; idx += 256)
      async_copy16(&lW[idx * 8], wsrc + (size_t)idx * 8);
  }

  if (tid < 128) { lsum[tid] = 0.f; lsq[tid] = 0.f; }
  if (GATHER) {
    if (tid < 64) pidx[tid] = gi[(size_t)g * KS + tid];
    if (tid < 3)  nx[tid] = new_xyz[(size_t)g * 3 + tid];
  } else {
    if (tid < 128) { lscale[tid] = scale[tid]; lshift[tid] = shift[tid]; }
  }
  __syncthreads();

  // ---- stage A (gather+center OR BN-apply+ReLU), convert to f16 ----
  if (GATHER) {
    for (int idx = tid; idx < 64 * KPAD; idx += 256) {
      const int r = idx / KPAD, c = idx - r * KPAD;
      float v = 0.f;
      const int p = pidx[r];
      if (c < 3)          v = xyz[((size_t)b * NPTS + p) * 3 + c] - nx[c];
      else if (c < KREAL) v = feat[(size_t)b * CFEAT * NPTS + (size_t)(c - 3) * NPTS + p];
      lA[idx] = (half_t)v;
    }
  } else {
    constexpr int CPR = KPAD / 8;  // 16B chunks per row
    for (int chunk = tid; chunk < 64 * CPR; chunk += 256) {
      const int r = chunk / CPR;
      const int c0 = (chunk - r * CPR) * 8;
      const v8h x = *(const v8h*)&yin[((size_t)g * KS + r) * KREAL + c0];
      v8h o;
      #pragma unroll
      for (int u = 0; u < 8; ++u)
        o[u] = (half_t)fmaxf((float)x[u] * lscale[c0 + u] + lshift[c0 + u], 0.f);
      *(v8h*)&lA[r * KPAD + c0] = o;
    }
  }
  wait_async0();     // W^T DMA landed (per-wave), then barrier orders LDS
  __syncthreads();

  const int wave  = tid >> 5, lane = tid & 31;
  const int m0    = (wave & 3) * 16;
  const int ntb   = (wave >> 2) * 4;
  const int lhalf = (lane < 16) ? 0 : 1;
  const int lrow  = lane & 15;

  v8f acc[4] = {};
  for (int ks = 0; ks < KPAD / 32; ++ks) {
    const int koff = ks * 32;
    // A fragment: lane holds row m0+lrow, K = g0..g0+15 (contiguous)
    v16h a;
    {
      const half_t* pa = &lA[(m0 + lrow) * KPAD + koff + lhalf * 16];
      #pragma unroll
      for (int j = 0; j < 8; ++j) { a[j] = pa[j]; a[8 + j] = pa[8 + j]; }
    }
    #pragma unroll
    for (int j = 0; j < 4; ++j) {
      // B fragment: lane holds col (ntb+j)*16+lrow, K = ks0..ks0+15 (contig in W^T)
      const int c = (ntb + j) * 16 + lrow;
      const half_t* pb = &lW[c * KPAD + koff + lhalf * 16];
      v16h bf;
      #pragma unroll
      for (int u = 0; u < 16; ++u) bf[u] = pb[u];
      acc[j] = __builtin_amdgcn_wmma_f32_16x16x32_f16(false, a, false, bf,
                                                      (short)0, acc[j], false, false);
    }
  }

  // ---- store y (f16) + per-channel sum/sumsq ----
  #pragma unroll
  for (int j = 0; j < 4; ++j) {
    const int cloc = (ntb + j) * 16 + lrow;
    float s = 0.f, s2 = 0.f;
    #pragma unroll
    for (int i = 0; i < 8; ++i) {
      const int row = m0 + (lhalf ? i + 8 : i);
      const float v = acc[j][i];
      yout[((size_t)g * KS + row) * NtotOut + n_off + cloc] = (half_t)v;
      s += v; s2 += v * v;
    }
    atomicAdd(&lsum[cloc], s);
    atomicAdd(&lsq[cloc], s2);
  }
  __syncthreads();
  if (tid < 128) {
    atomicAdd(&gsum[n_off + tid], lsum[tid]);
    atomicAdd(&gsq[n_off + tid], lsq[tid]);
  }
}

// =====================================================================
// 5) BN3+ReLU, max over K=64, transposed store (B,256,1024)
// =====================================================================
__global__ __launch_bounds__(256) void final_max(const half_t* __restrict__ y3,
                                                 const float* __restrict__ scale,
                                                 const float* __restrict__ shift,
                                                 float* __restrict__ out) {
  const int g = blockIdx.x, c = threadIdx.x;
  const int b = g >> 10, s = g & 1023;
  const half_t* base = y3 + (size_t)g * KS * 256 + c;
  const float sc = scale[c], sh = shift[c];
  float best = -1e30f;
  #pragma unroll 4
  for (int k = 0; k < KS; ++k) {
    const float v = (float)base[(size_t)k * 256];
    best = fmaxf(best, fmaxf(v * sc + sh, 0.f));
  }
  out[24576 + ((size_t)b * 256 + (size_t)c) * SPTS + s] = best;
}

// =====================================================================
extern "C" void kernel_launch(void* const* d_in, const int* in_sizes, int n_in,
                              void* d_out, int out_size, void* d_ws, size_t ws_size,
                              hipStream_t stream) {
  (void)in_sizes; (void)n_in; (void)out_size; (void)ws_size;
  const float* xyz  = (const float*)d_in[0];
  const float* feat = (const float*)d_in[1];
  const float* W1 = (const float*)d_in[2];
  const float* g1 = (const float*)d_in[3];
  const float* b1 = (const float*)d_in[4];
  const float* W2 = (const float*)d_in[5];
  const float* g2 = (const float*)d_in[6];
  const float* b2 = (const float*)d_in[7];
  const float* W3 = (const float*)d_in[8];
  const float* g3 = (const float*)d_in[9];
  const float* b3 = (const float*)d_in[10];

  float* out = (float*)d_out;
  char*  ws  = (char*)d_ws;

  int*    gi    = (int*)(ws + OFF_GI);
  float*  stats = (float*)(ws + OFF_STATS);
  float *sum1 = stats,        *sq1 = stats + 128;
  float *sum2 = stats + 256,  *sq2 = stats + 384;
  float *sum3 = stats + 512,  *sq3 = stats + 768;
  float *sc1  = stats + 1024, *sh1 = stats + 1152;
  float *sc2  = stats + 1280, *sh2 = stats + 1408;
  float *sc3  = stats + 1536, *sh3 = stats + 1792;
  half_t* wt1 = (half_t*)(ws + OFF_WT1);
  half_t* wt2 = (half_t*)(ws + OFF_WT2);
  half_t* wt3 = (half_t*)(ws + OFF_WT3);
  half_t* y1 = (half_t*)(ws + OFF_Y1);
  half_t* y2 = (half_t*)(ws + OFF_Y2);
  half_t* y3 = (half_t*)(ws + OFF_Y3);
  float* new_xyz = out;  // first 8*1024*3 floats of output

  fps_kernel<<<BATCH, 512, 0, stream>>>(xyz, new_xyz);
  qbp_kernel<<<NGROUP / 8, 256, 0, stream>>>(xyz, new_xyz, gi);
  zero_stats<<<8, 256, 0, stream>>>(stats, 2048);
  prep_w<<<(128 * 160 + 255) / 256, 256, 0, stream>>>(W1, wt1, 131, 160, 128);
  prep_w<<<(128 * 128 + 255) / 256, 256, 0, stream>>>(W2, wt2, 128, 128, 128);
  prep_w<<<(256 * 128 + 255) / 256, 256, 0, stream>>>(W3, wt3, 128, 128, 256);

  // layer 1: gather/concat fused, K=131 padded to 160
  gemm_bn_kernel<160, 131, true><<<dim3(NGROUP, 1), 256, 0, stream>>>(
      xyz, feat, new_xyz, gi, nullptr, nullptr, nullptr, wt1, 128, y1, sum1, sq1);
  bn_finalize<<<1, 128, 0, stream>>>(sum1, sq1, g1, b1, sc1, sh1, 128);

  // layer 2: BN1+ReLU fused into A staging
  gemm_bn_kernel<128, 128, false><<<dim3(NGROUP, 1), 256, 0, stream>>>(
      xyz, feat, new_xyz, gi, y1, sc1, sh1, wt2, 128, y2, sum2, sq2);
  bn_finalize<<<1, 128, 0, stream>>>(sum2, sq2, g2, b2, sc2, sh2, 128);

  // layer 3: N=256 split over gridDim.y
  gemm_bn_kernel<128, 128, false><<<dim3(NGROUP, 2), 256, 0, stream>>>(
      xyz, feat, new_xyz, gi, y2, sc2, sh2, wt3, 256, y3, sum3, sq3);
  bn_finalize<<<1, 256, 0, stream>>>(sum3, sq3, g3, b3, sc3, sh3, 256);

  final_max<<<NGROUP, 256, 0, stream>>>(y3, sc3, sh3, out);
}
